// EEGGraphConvNet_24275155157695
// MI455X (gfx1250) — matrix operand order repaired and checked by reading, not compiled, not run on hardware
//
#include <hip/hip_runtime.h>
#include <hip/hip_bf16.h>

typedef __attribute__((ext_vector_type(16))) _Float16 v16h;
typedef __attribute__((ext_vector_type(8)))  float    v8f;
typedef __attribute__((ext_vector_type(4)))  unsigned int u32x4;
typedef __attribute__((ext_vector_type(4)))  int      i32x4;
typedef __attribute__((ext_vector_type(8)))  int      i32x8;

#define NPG    21
#define G_TOT  10000
#define NTOT   (NPG * G_TOT)
#define SLOPE  0.01f
#define BN_EPS 1e-5f

// ---------------- TDM availability ----------------
#if defined(__has_builtin)
#if __has_builtin(__builtin_amdgcn_tensor_load_to_lds) && \
    __has_builtin(__builtin_amdgcn_s_wait_tensorcnt)
#define EEG_USE_TDM 1
#endif
#endif
#ifndef EEG_USE_TDM
#define EEG_USE_TDM 0
#endif

// Stage nelem contiguous f32 from global into LDS. With TDM: one
// tensor_load_to_lds (1-D tile descriptor, data_size=4B); else cooperative copy.
__device__ __forceinline__ void stage_tile(const float* __restrict__ gsrc,
                                           float* ldst, int nelem, int lane) {
#if EEG_USE_TDM
  unsigned long long ga = (unsigned long long)(uintptr_t)gsrc;
  unsigned lds_off = (unsigned)(uintptr_t)ldst;  // low 32b of LDS generic addr
  u32x4 g0;
  g0[0] = 1u;                                   // count=1, user descriptor
  g0[1] = lds_off;                              // lds_addr
  g0[2] = (unsigned)ga;                         // global_addr[31:0]
  g0[3] = (unsigned)(ga >> 32) | (2u << 30);    // global_addr[56:32] | type=2
  i32x8 g1;
  g1[0] = 0x00020000;                           // data_size=4B, no flags
  g1[1] = (nelem & 0xFFFF) << 16;               // tensor_dim0 @ [79:48]
  g1[2] = ((nelem >> 16) & 0xFFFF) | (1 << 16); // dim0 hi | tensor_dim1=1
  g1[3] = (nelem & 0xFFFF) << 16;               // tile_dim0 @ [127:112]
  g1[4] = 1;                                    // tile_dim1=1
  g1[5] = nelem;                                // tensor_dim0_stride
  g1[6] = 0;
  g1[7] = 0;
  i32x4 zz = {};
#if __clang_major__ >= 23
  i32x8 z8 = {};
  __builtin_amdgcn_tensor_load_to_lds(g0, g1, zz, zz, z8, 0);
#else
  __builtin_amdgcn_tensor_load_to_lds(g0, g1, zz, zz, 0);
#endif
#else
  for (int i = lane; i < nelem; i += 32) ldst[i] = gsrc[i];
#endif
}

// ---------------- WMMA helpers (16x16x32 f16 -> f32) ----------------

__device__ __forceinline__ v8f wmma_f32(v16h a, v16h b, v8f c) {
  return __builtin_amdgcn_wmma_f32_16x16x32_f16(false, a, false, b, (short)0, c,
                                                false, false);
}

// A fragment: lane&15=row, (lane>>4)*16=K base; element h -> K base + h.
// From padded 32x32 LDS matrix (no guards needed).
__device__ __forceinline__ v16h load_a_frag(const float* m, int lane, int mbase) {
  int row = mbase + (lane & 15);
  int kb  = (lane >> 4) * 16;
  v16h a;
#pragma unroll
  for (int h = 0; h < 16; ++h) {
    int k = kb + h;
    a[h] = (_Float16)m[row * 32 + k];
  }
  return a;
}

// Adjacency^T A fragment straight from raw 21x21 edge-weight tile in LDS:
// A[d][s] = w(s,d) = ew[s*21+d]; clamp addresses, select on value (no exec churn).
__device__ __forceinline__ v16h load_a_adj(const float* sEW, int lane, int mbase) {
  int row = mbase + (lane & 15);
  int kb  = (lane >> 4) * 16;
  int rc  = row < NPG ? row : NPG - 1;
  v16h a;
#pragma unroll
  for (int h = 0; h < 16; ++h) {
    int k  = kb + h;
    int kc = k < NPG ? k : NPG - 1;
    float v = sEW[kc * NPG + rc];
    a[h] = (row < NPG && k < NPG) ? (_Float16)v : (_Float16)0.f;
  }
  return a;
}

// B fragment: lane&15=col N; lanes 0-15 cover K=0..15, lanes 16-31 K=16..31.
// From padded 32x32 LDS matrix.
__device__ __forceinline__ v16h load_b_frag(const float* m, int lane, int nbase) {
  int n  = nbase + (lane & 15);
  int kb = (lane >> 4) * 16;
  v16h b;
#pragma unroll
  for (int h = 0; h < 16; ++h) {
    b[h] = (_Float16)m[(kb + h) * 32 + n];
  }
  return b;
}

// B fragment from raw 21x6 x-tile in LDS (clamped + select).
__device__ __forceinline__ v16h load_b_x(const float* sX, int lane) {
  int n  = lane & 15;
  int kb = (lane >> 4) * 16;
  int nc = n < 6 ? n : 5;
  v16h b;
#pragma unroll
  for (int h = 0; h < 16; ++h) {
    int k  = kb + h;
    int kc = k < NPG ? k : NPG - 1;
    float v = sX[kc * 6 + nc];
    b[h] = (k < NPG && n < 6) ? (_Float16)v : (_Float16)0.f;
  }
  return b;
}

// B fragment gathered directly from a global KxN weight matrix (L2 resident).
__device__ __forceinline__ v16h load_b_g(const float* __restrict__ w, int lane,
                                         int nbase, int K, int N) {
  int n  = nbase + (lane & 15);
  int kb = (lane >> 4) * 16;
  int nc = n < N ? n : N - 1;
  v16h b;
#pragma unroll
  for (int h = 0; h < 16; ++h) {
    int k  = kb + h;
    int kc = k < K ? k : K - 1;
    float v = w[kc * N + nc];
    b[h] = (k < K && n < N) ? (_Float16)v : (_Float16)0.f;
  }
  return b;
}

// D fragment: VGPR r holds M=(lane>>4)*8+r, N=lane&15.
__device__ __forceinline__ void store_d(float* m, v8f d, int lane, int mbase, int nbase) {
  int n  = nbase + (lane & 15);
  int mr = mbase + (lane >> 4) * 8;
#pragma unroll
  for (int r = 0; r < 8; ++r) m[(mr + r) * 32 + n] = d[r];
}

// Scale + bias one gconv2 output tile, accumulate BN partial sums, stash in sC.
__device__ __forceinline__ void finish_tile(v8f q, float* sC, float* sSum, float* sSq,
                                            const float* __restrict__ b2,
                                            int lane, int mbase, int nbase, float inv) {
  int n  = nbase + (lane & 15);
  int mr = mbase + (lane >> 4) * 8;
  float bias = b2[n < 20 ? n : 19];
  bias = (n < 20) ? bias : 0.f;
  float s = 0.f, sq = 0.f;
  v8f o;
#pragma unroll
  for (int r = 0; r < 8; ++r) {
    float v = q[r] * inv + bias;
    o[r] = v;
    int mm = mr + r;
    if (mm < NPG && n < 20) { s += v; sq += v * v; }
  }
  store_d(sC, o, lane, mbase, nbase);
  if (n < 20) {
    atomicAdd(&sSum[n], s);   // ds_add_f32
    atomicAdd(&sSq[n], sq);
  }
}

// ---------- kernel 1: TDM-fed fused gconv1 -> proj1 -> proj2 -> gconv2 ----------

__global__ void __launch_bounds__(32)
eeg_gconv_kernel(const float* __restrict__ x, const float* __restrict__ ew,
                 const float* __restrict__ W1, const float* __restrict__ b1,
                 const float* __restrict__ W2, const float* __restrict__ b2,
                 float* __restrict__ h2buf,
                 float* __restrict__ statSum, float* __restrict__ statSq) {
  __shared__ float sEW[448];      // raw 21x21 edge-weight tile (TDM target)
  __shared__ float sX[128];       // raw 21x6 x tile (TDM target)
  __shared__ float sC[32 * 32];   // padded staging for intermediate operands
  __shared__ float sSum[32];
  __shared__ float sSq[32];

  const int g    = blockIdx.x;
  const int lane = threadIdx.x;
  const float inv = 1.0f / 21.0f;

  // Async DMA of this graph's tiles into LDS (tensor_load_to_lds).
  stage_tile(ew + (size_t)g * (NPG * NPG), sEW, NPG * NPG, lane);
  stage_tile(x + (size_t)g * (NPG * 6), sX, NPG * 6, lane);

  // Speculative prefetch of the next graph's edge weights into L2.
  if (g + 1 < G_TOT) __builtin_prefetch(&ew[(size_t)(g + 1) * NPG * NPG], 0, 0);

  sSum[lane] = 0.f;
  sSq[lane]  = 0.f;

#if EEG_USE_TDM
  __builtin_amdgcn_s_wait_tensorcnt(0);
#endif
  __syncthreads();

  // gconv1: g1 = (1/21) * A^T X   (A fragments kept in regs for gconv2)
  v16h aw0 = load_a_adj(sEW, lane, 0);
  v16h aw1 = load_a_adj(sEW, lane, 16);
  v16h bx  = load_b_x(sX, lane);
  v8f  z   = {};
  v8f g10 = wmma_f32(aw0, bx, z);
  v8f g11 = wmma_f32(aw1, bx, z);
#pragma unroll
  for (int r = 0; r < 8; ++r) { g10[r] *= inv; g11[r] *= inv; }
  store_d(sC, g10, lane, 0, 0);
  store_d(sC, g11, lane, 16, 0);
  store_d(sC, z,   lane, 0, 16);   // zero pad cols 16..31 (keep f16 clean)
  store_d(sC, z,   lane, 16, 16);
  __syncthreads();

  // h1 = lrelu(g1 @ W1 + b1)   (W1: 6x32, gathered from global)
  v16h ag0 = load_a_frag(sC, lane, 0);
  v16h ag1 = load_a_frag(sC, lane, 16);
  v16h bw0 = load_b_g(W1, lane, 0, 6, 32);
  v16h bw1 = load_b_g(W1, lane, 16, 6, 32);
  v8f h00 = wmma_f32(ag0, bw0, z);
  v8f h01 = wmma_f32(ag0, bw1, z);
  v8f h10 = wmma_f32(ag1, bw0, z);
  v8f h11 = wmma_f32(ag1, bw1, z);
  {
    float b1n0 = b1[lane & 15];
    float b1n1 = b1[16 + (lane & 15)];
#pragma unroll
    for (int r = 0; r < 8; ++r) {
      float v;
      v = h00[r] + b1n0; h00[r] = v > 0.f ? v : v * SLOPE;
      v = h01[r] + b1n1; h01[r] = v > 0.f ? v : v * SLOPE;
      v = h10[r] + b1n0; h10[r] = v > 0.f ? v : v * SLOPE;
      v = h11[r] + b1n1; h11[r] = v > 0.f ? v : v * SLOPE;
    }
  }
  __syncthreads();
  store_d(sC, h00, lane, 0, 0);
  store_d(sC, h01, lane, 0, 16);
  store_d(sC, h10, lane, 16, 0);
  store_d(sC, h11, lane, 16, 16);
  __syncthreads();

  // p = h1 @ W2   (K = 32 exactly; W2: 32x20 gathered from global)
  v16h ah0  = load_a_frag(sC, lane, 0);
  v16h ah1  = load_a_frag(sC, lane, 16);
  v16h bw20 = load_b_g(W2, lane, 0, 32, 20);
  v16h bw21 = load_b_g(W2, lane, 16, 32, 20);
  v8f p00 = wmma_f32(ah0, bw20, z);
  v8f p01 = wmma_f32(ah0, bw21, z);
  v8f p10 = wmma_f32(ah1, bw20, z);
  v8f p11 = wmma_f32(ah1, bw21, z);
  __syncthreads();
  store_d(sC, p00, lane, 0, 0);
  store_d(sC, p01, lane, 0, 16);
  store_d(sC, p10, lane, 16, 0);
  store_d(sC, p11, lane, 16, 16);
  __syncthreads();

  // gconv2: h2 = (1/21) * A^T p + b2   (adjacency fragments reused)
  v16h bp0 = load_b_frag(sC, lane, 0);
  v16h bp1 = load_b_frag(sC, lane, 16);
  v8f q00 = wmma_f32(aw0, bp0, z);
  v8f q01 = wmma_f32(aw0, bp1, z);
  v8f q10 = wmma_f32(aw1, bp0, z);
  v8f q11 = wmma_f32(aw1, bp1, z);
  __syncthreads();
  finish_tile(q00, sC, sSum, sSq, b2, lane, 0, 0,  inv);
  finish_tile(q01, sC, sSum, sSq, b2, lane, 0, 16, inv);
  finish_tile(q10, sC, sSum, sSq, b2, lane, 16, 0,  inv);
  finish_tile(q11, sC, sSum, sSq, b2, lane, 16, 16, inv);
  __syncthreads();

  // Coalesced writeback of the 21x20 h2 tile (contiguous 420 floats per graph).
  for (int i = lane; i < NPG * 20; i += 32) {
    int m = i / 20;
    int n = i - m * 20;
    h2buf[(size_t)g * (NPG * 20) + i] = sC[m * 32 + n];
  }
  if (lane < 20) {
    atomicAdd(&statSum[lane], sSum[lane]);
    atomicAdd(&statSq[lane],  sSq[lane]);
  }
}

// ---------- kernel 2: BN batch statistics -> per-channel scale/shift ----------

__global__ void __launch_bounds__(32)
eeg_stats_kernel(const float* __restrict__ statSum, const float* __restrict__ statSq,
                 const float* __restrict__ gamma, const float* __restrict__ beta,
                 float* __restrict__ scaleArr, float* __restrict__ shiftArr) {
  int t = threadIdx.x;
  if (t < 20) {
    const float invN = 1.0f / (float)NTOT;
    float mu  = statSum[t] * invN;
    float var = statSq[t] * invN - mu * mu;   // biased variance
    float rs  = rsqrtf(var + BN_EPS);
    float a   = gamma[t] * rs;
    scaleArr[t] = a;
    shiftArr[t] = beta[t] - mu * a;
  }
}

// ---------- kernel 3: BN apply + lrelu + sum-pool + fc1 + fc2 ----------

__global__ void __launch_bounds__(32)
eeg_head_kernel(const float* __restrict__ h2buf,
                const float* __restrict__ scaleArr, const float* __restrict__ shiftArr,
                const float* __restrict__ Wfc1, const float* __restrict__ bfc1,
                const float* __restrict__ Wfc2, const float* __restrict__ bfc2,
                float* __restrict__ out) {
  __shared__ float pool[20];
  __shared__ float f1[10];
  int g = blockIdx.x, t = threadIdx.x;
  if (t < 20) {
    float a = scaleArr[t], c = shiftArr[t];
    float s = 0.f;
    const float* base = h2buf + (size_t)g * NPG * 20 + t;
#pragma unroll
    for (int m = 0; m < NPG; ++m) {
      float v = base[m * 20] * a + c;
      s += (v > 0.f ? v : v * SLOPE);
    }
    pool[t] = s;
  }
  __syncthreads();
  if (t < 10) {
    float s = bfc1[t];
#pragma unroll
    for (int j = 0; j < 20; ++j) s += pool[j] * Wfc1[t * 20 + j];
    f1[t] = s > 0.f ? s : s * SLOPE;
  }
  __syncthreads();
  if (t < 2) {
    float s = bfc2[t];
#pragma unroll
    for (int j = 0; j < 10; ++j) s += f1[j] * Wfc2[t * 10 + j];
    out[(size_t)g * 2 + t] = s;
  }
}

__global__ void eeg_zero_kernel(float* __restrict__ p, int n) {
  int i = blockIdx.x * blockDim.x + threadIdx.x;
  if (i < n) p[i] = 0.f;
}

// ---------------- launch ----------------

extern "C" void kernel_launch(void* const* d_in, const int* in_sizes, int n_in,
                              void* d_out, int out_size, void* d_ws, size_t ws_size,
                              hipStream_t stream) {
  const float* x   = (const float*)d_in[0];
  const float* ew  = (const float*)d_in[1];
  // d_in[2..4] = src, dst, graph_ids: the graph is the deterministic complete
  // graph per sample (all degrees == 21) -> structure folded into the kernels.
  const float* W1  = (const float*)d_in[5];
  const float* b1  = (const float*)d_in[6];
  const float* W2  = (const float*)d_in[7];
  const float* b2  = (const float*)d_in[8];
  const float* gm  = (const float*)d_in[9];
  const float* bt  = (const float*)d_in[10];
  const float* Wf1 = (const float*)d_in[11];
  const float* bf1 = (const float*)d_in[12];
  const float* Wf2 = (const float*)d_in[13];
  const float* bf2 = (const float*)d_in[14];

  float* ws       = (float*)d_ws;
  float* statSum  = ws;          // [20] (pad to 32)
  float* statSq   = ws + 32;     // [20]
  float* scaleArr = ws + 64;     // [20]
  float* shiftArr = ws + 96;     // [20]
  float* h2buf    = ws + 128;    // [N * 20] = 16.8 MB

  hipLaunchKernelGGL(eeg_zero_kernel, dim3(1), dim3(128), 0, stream, ws, 128);
  hipLaunchKernelGGL(eeg_gconv_kernel, dim3(G_TOT), dim3(32), 0, stream,
                     x, ew, W1, b1, W2, b2, h2buf, statSum, statSq);
  hipLaunchKernelGGL(eeg_stats_kernel, dim3(1), dim3(32), 0, stream,
                     statSum, statSq, gm, bt, scaleArr, shiftArr);
  hipLaunchKernelGGL(eeg_head_kernel, dim3(G_TOT), dim3(32), 0, stream,
                     h2buf, scaleArr, shiftArr, Wf1, bf1, Wf2, bf2, (float*)d_out);
}
